// GATv2Model_8761733284460
// MI455X (gfx1250) — matrix operand order repaired and checked by reading, not compiled, not run on hardware
//
#include <hip/hip_runtime.h>

typedef __attribute__((ext_vector_type(16))) _Float16 v16h;
typedef __attribute__((ext_vector_type(8)))  _Float16 v8h;
typedef __attribute__((ext_vector_type(8)))  float    v8f;
typedef __attribute__((ext_vector_type(4)))  unsigned int v4u;
typedef __attribute__((ext_vector_type(4)))  int v4i;
typedef __attribute__((ext_vector_type(8)))  int v8i;

// Tensor Data Mover path: ROCm 7.2 (clang-22) 5-arg builtin only; clang-23+
// has a different arity -> fall back to manual staging there.
#if defined(__HIP_DEVICE_COMPILE__) && __has_builtin(__builtin_amdgcn_tensor_load_to_lds) && (__clang_major__ < 23)
#define USE_TDM 1
#else
#define USE_TDM 0
#endif

// Report which path the DEVICE pass took (visible in compiler stderr).
#if defined(__HIP_DEVICE_COMPILE__)
# if USE_TDM
#  pragma message("CDNA5 device pass: USE_TDM=1 (tensor_load_to_lds 5-arg builtin active)")
# else
#  pragma message("CDNA5 device pass: USE_TDM=0 (manual LDS staging fallback)")
# endif
#endif

namespace {

constexpr int kBsz = 16, kN = 128, kHid = 128, kH = 8, kC = 16, kL = 4;
constexpr float kNegSlope = 0.2f, kEps = 1e-5f;
constexpr int kMat = kHid * kHid;   // 16384 elements per 128x128 matrix

// ---------------------------------------------------------------------------
// TDM: flat contiguous copy global -> LDS, expressed as a 1-D tensor tile of
// 8-byte elements (D# layout per CDNA5 ISA).
__device__ __forceinline__ void tdm_copy_to_lds(void* lds_dst, const void* gsrc,
                                                unsigned bytes) {
#if USE_TDM
  unsigned lds_addr = (unsigned)(unsigned long long)(uintptr_t)lds_dst; // addr[31:0] = LDS offset
  unsigned long long ga = (unsigned long long)(uintptr_t)gsrc;
  unsigned elems = bytes >> 3;                       // data_size = 8B units
  v4u g0;
  g0[0] = 1u;                                        // count=1, user descriptor
  g0[1] = lds_addr;                                  // lds_addr [63:32]
  g0[2] = (unsigned)ga;                              // global_addr [95:64]
  g0[3] = (unsigned)((ga >> 32) & 0x01FFFFFFu) | (2u << 30); // addr hi + type=2
  v8i g1;
  g1[0] = (int)(3u << 16);                           // data_size=3 (8 bytes)
  g1[1] = (int)((elems & 0xFFFFu) << 16);            // tensor_dim0[15:0]  -> bits[63:48]
  g1[2] = (int)((elems >> 16) & 0xFFFFu);            // tensor_dim0[31:16] -> bits[79:64]
  g1[3] = (int)((elems & 0xFFFFu) << 16);            // tile_dim0 -> bits[127:112]
  g1[4] = 0;                                         // tile_dim1/2 unused
  g1[5] = (int)elems;                                // tensor_dim0_stride[31:0]
  g1[6] = 0;
  g1[7] = 0;
  v4i gz = {0, 0, 0, 0};
  __builtin_amdgcn_tensor_load_to_lds(g0, g1, gz, gz, 0);
#else
  (void)lds_dst; (void)gsrc; (void)bytes;
#endif
}

__device__ __forceinline__ void tdm_wait() {
#if USE_TDM
  __builtin_amdgcn_s_wait_tensorcnt(0);
#endif
}

// ---------------------------------------------------------------------------
// WMMA fragment loaders (layouts per CDNA5 ISA 7.12.2, 16-bit A 16x32 / B 32x16)
__device__ __forceinline__ v16h frag_a_from_lds(const _Float16* base, int k0) {
  int lane = threadIdx.x & 31;
  int row  = lane & 15;
  int kb   = k0 + ((lane & 16) ? 8 : 0);
  const _Float16* p = base + row * 128 + kb;
  v8h lo = *(const v8h*)(p);
  v8h hi = *(const v8h*)(p + 16);
  v16h r;
#pragma unroll
  for (int e = 0; e < 8; ++e) { r[e] = lo[e]; r[e + 8] = hi[e]; }
  return r;
}

__device__ __forceinline__ v16h frag_b_from_lds_T(const _Float16* baseT, int k0) {
  int lane = threadIdx.x & 31;
  int col  = lane & 15;
  int kb   = k0 + ((lane & 16) ? 8 : 0);
  const _Float16* p = baseT + col * 128 + kb;
  v8h lo = *(const v8h*)(p);
  v8h hi = *(const v8h*)(p + 16);
  v16h r;
#pragma unroll
  for (int e = 0; e < 8; ++e) { r[e] = lo[e]; r[e + 8] = hi[e]; }
  return r;
}

__device__ __forceinline__ float block_sum(float v, float* smem) {
  __syncthreads();
#pragma unroll
  for (int m = 16; m > 0; m >>= 1) v += __shfl_xor(v, m, 32);
  int wid = threadIdx.x >> 5;
  if ((threadIdx.x & 31) == 0) smem[wid] = v;
  __syncthreads();
  if (threadIdx.x == 0) {
    float s = 0.f;
    int nw = blockDim.x >> 5;
    for (int i = 0; i < nw; ++i) s += smem[i];
    smem[0] = s;
  }
  __syncthreads();
  return smem[0];
}

__device__ __forceinline__ float block_max(float v, float* smem) {
  __syncthreads();
#pragma unroll
  for (int m = 16; m > 0; m >>= 1) v = fmaxf(v, __shfl_xor(v, m, 32));
  int wid = threadIdx.x >> 5;
  if ((threadIdx.x & 31) == 0) smem[wid] = v;
  __syncthreads();
  if (threadIdx.x == 0) {
    float s = -3.4e38f;
    int nw = blockDim.x >> 5;
    for (int i = 0; i < nw; ++i) s = fmaxf(s, smem[i]);
    smem[0] = s;
  }
  __syncthreads();
  return smem[0];
}

} // namespace

// ---------------------------------------------------------------------------
__global__ void k_cvt_f16(const float* __restrict__ src, _Float16* __restrict__ dst, int n) {
  int i = blockIdx.x * blockDim.x + threadIdx.x;
  if (i < n) dst[i] = (_Float16)src[i];
}

__global__ void k_gather_embed(const float* __restrict__ emb, const int* __restrict__ cat,
                               _Float16* __restrict__ dst) {
  int i = blockIdx.x * blockDim.x + threadIdx.x;   // over N*N*HID exactly
  int pair = i >> 7, c = i & 127;
  dst[i] = (_Float16)emb[cat[pair] * kHid + c];
}

// Input MLP stage 1: t = relu(LN(x @ w1 + b1)) ; one block per (b,n) row.
__global__ void k_input_mlp1(const float* __restrict__ x, const float* __restrict__ w1,
                             const float* __restrict__ b1, const float* __restrict__ g,
                             const float* __restrict__ bb, _Float16* __restrict__ t16) {
  __shared__ float red[8];
  int row = blockIdx.x, c = threadIdx.x;
  float x0 = x[row * 2 + 0], x1 = x[row * 2 + 1];
  float t = fmaf(x0, w1[c], fmaf(x1, w1[kHid + c], b1[c]));
  float mu  = block_sum(t, red) * (1.f / kHid);
  float d   = t - mu;
  float var = block_sum(d * d, red) * (1.f / kHid);
  float y = d * rsqrtf(var + kEps) * g[c] + bb[c];
  t16[row * kHid + c] = (_Float16)fmaxf(y, 0.f);
}

// ---------------------------------------------------------------------------
// WMMA GEMM: out[M,128] = X16[M,128] @ W16[128,128] (+bias).
// Block covers MT*16 rows; wave w owns output cols [16w,16w+16) and MT row
// tiles, so each B fragment is reused MT times (4*MT wmma per wave).
template <int MT, bool HasBias, bool OutF, bool OutH>
__global__ void __launch_bounds__(256) k_gemm(const _Float16* __restrict__ X,
                                              const _Float16* __restrict__ W,
                                              const float* __restrict__ bias,
                                              float* __restrict__ outF,
                                              _Float16* __restrict__ outH) {
  __shared__ _Float16 sA[MT * 16 * 128];
  __shared__ _Float16 sWt[128 * 128];   // transposed: [n][k]
  int tid = threadIdx.x;
  int m0  = blockIdx.x * (MT * 16);
  __builtin_prefetch(W + (tid << 6), 0, 3);           // global_prefetch_b8
#if USE_TDM
  if ((tid >> 5) == 0) tdm_copy_to_lds(sA, X + m0 * 128, MT * 16 * 128 * 2);
#else
  for (int idx = tid; idx < MT * 16 * 128; idx += 256) sA[idx] = X[m0 * 128 + idx];
#endif
  for (int idx = tid; idx < 128 * 128; idx += 256) {
    int k = idx >> 7, n = idx & 127;
    sWt[n * 128 + k] = W[idx];
  }
  tdm_wait();
  __syncthreads();
  int w = tid >> 5;
  const v8f zero = {0.f, 0.f, 0.f, 0.f, 0.f, 0.f, 0.f, 0.f};
  v8f acc[MT];
#pragma unroll
  for (int mt = 0; mt < MT; ++mt) acc[mt] = zero;
#pragma unroll
  for (int ks = 0; ks < 4; ++ks) {
    v16h b = frag_b_from_lds_T(sWt + (w * 16) * 128, ks * 32);
#pragma unroll
    for (int mt = 0; mt < MT; ++mt) {
      v16h a = frag_a_from_lds(sA + mt * (16 * 128), ks * 32);
      acc[mt] = __builtin_amdgcn_wmma_f32_16x16x32_f16(false, a, false, b,
                                                       (short)0, acc[mt], false, false);
    }
  }
  int lane = tid & 31;
  int col  = w * 16 + (lane & 15);
  int roff = (lane & 16) ? 8 : 0;        // C/D layout: lanes 16-31 hold M=8..15
  float bv = HasBias ? bias[col] : 0.f;
#pragma unroll
  for (int mt = 0; mt < MT; ++mt) {
#pragma unroll
    for (int r = 0; r < 8; ++r) {
      int row = m0 + mt * 16 + r + roff;
      float v = acc[mt][r] + bv;
      if constexpr (OutF) outF[row * 128 + col] = v;
      if constexpr (OutH) outH[row * 128 + col] = (_Float16)v;
    }
  }
}

// WMMA GEMM + bias + row-LayerNorm (+relu, +residual) fused.
template <int MT, bool Relu, bool HasRes>
__global__ void __launch_bounds__(256) k_gemm_ln(const _Float16* __restrict__ X,
                                                 const _Float16* __restrict__ W,
                                                 const float* __restrict__ bias,
                                                 const float* __restrict__ g,
                                                 const float* __restrict__ lb,
                                                 const float* __restrict__ residual,
                                                 float* __restrict__ outF,
                                                 _Float16* __restrict__ outH) {
  __shared__ _Float16 sA[MT * 16 * 128];
  __shared__ _Float16 sWt[128 * 128];
  __shared__ float sO[MT * 16 * 128];
  __shared__ float sMu[MT * 16], sRs[MT * 16];
  int tid = threadIdx.x;
  int m0  = blockIdx.x * (MT * 16);
  __builtin_prefetch(W + (tid << 6), 0, 3);
#if USE_TDM
  if ((tid >> 5) == 0) tdm_copy_to_lds(sA, X + m0 * 128, MT * 16 * 128 * 2);
#else
  for (int idx = tid; idx < MT * 16 * 128; idx += 256) sA[idx] = X[m0 * 128 + idx];
#endif
  for (int idx = tid; idx < 128 * 128; idx += 256) {
    int k = idx >> 7, n = idx & 127;
    sWt[n * 128 + k] = W[idx];
  }
  tdm_wait();
  __syncthreads();
  int w = tid >> 5;
  const v8f zero = {0.f, 0.f, 0.f, 0.f, 0.f, 0.f, 0.f, 0.f};
  v8f acc[MT];
#pragma unroll
  for (int mt = 0; mt < MT; ++mt) acc[mt] = zero;
#pragma unroll
  for (int ks = 0; ks < 4; ++ks) {
    v16h b = frag_b_from_lds_T(sWt + (w * 16) * 128, ks * 32);
#pragma unroll
    for (int mt = 0; mt < MT; ++mt) {
      v16h a = frag_a_from_lds(sA + mt * (16 * 128), ks * 32);
      acc[mt] = __builtin_amdgcn_wmma_f32_16x16x32_f16(false, a, false, b,
                                                       (short)0, acc[mt], false, false);
    }
  }
  int lane = tid & 31;
  int col  = w * 16 + (lane & 15);
  int roff = (lane & 16) ? 8 : 0;
  float bv = bias[col];
#pragma unroll
  for (int mt = 0; mt < MT; ++mt)
#pragma unroll
    for (int r = 0; r < 8; ++r)
      sO[(mt * 16 + r + roff) * 128 + col] = acc[mt][r] + bv;
  __syncthreads();
  if (tid < MT * 16) {
    float s = 0.f, s2 = 0.f;
    for (int c = 0; c < 128; ++c) { float v = sO[tid * 128 + c]; s += v; s2 += v * v; }
    float mu  = s * (1.f / 128.f);
    float var = s2 * (1.f / 128.f) - mu * mu;
    sMu[tid] = mu; sRs[tid] = rsqrtf(var + kEps);
  }
  __syncthreads();
  for (int idx = tid; idx < MT * 16 * 128; idx += 256) {
    int r = idx >> 7, c = idx & 127;
    float v = (sO[idx] - sMu[r]) * sRs[r] * g[c] + lb[c];
    if constexpr (Relu) v = fmaxf(v, 0.f);
    int grow = m0 + r;
    if constexpr (HasRes) v += residual[grow * 128 + c];
    outF[grow * 128 + c] = v;
    outH[grow * 128 + c] = (_Float16)v;
  }
}

// ---------------------------------------------------------------------------
// Attention scores: s[b,j,h,i] = sum_c leaky(xl[b,i,hc]+xr[b,j,hc]+e[i,j,hc])*att[hc]
__global__ void __launch_bounds__(256) k_score(const float* __restrict__ xl,
                                               const float* __restrict__ xr,
                                               const float* __restrict__ e32,
                                               const float* __restrict__ att,
                                               float* __restrict__ s) {
  __shared__ float sXr[128], sAtt[128];
  int bj = blockIdx.x;
  int b = bj >> 7, j = bj & 127;
  int tid = threadIdx.x;
  if (tid < 128) { sXr[tid] = xr[bj * 128 + tid]; sAtt[tid] = att[tid]; }
  __syncthreads();
  int half = tid >> 7, hc = tid & 127, lane = tid & 31;
  for (int ib = 0; ib < 128; ib += 2) {
    int i = ib + half;
    float v = xl[(b * 128 + i) * 128 + hc] + sXr[hc] + e32[(i * 128 + j) * 128 + hc];
    float lv = v > 0.f ? v : kNegSlope * v;
    float p = lv * sAtt[hc];
#pragma unroll
    for (int m = 1; m < 16; m <<= 1) p += __shfl_xor(p, m, 32);  // reduce over C=16
    if ((lane & 15) == 0) s[(bj * kH + (hc >> 4)) * 128 + i] = p;
  }
}

// Softmax over source i; writes a16 in [b,h,j,i] layout for WMMA aggregation.
__global__ void k_softmax(const float* __restrict__ s, _Float16* __restrict__ a16) {
  __shared__ float red[8];
  int blk = blockIdx.x;          // (b*N + j)*H + h
  int tid = threadIdx.x;
  float v = s[blk * 128 + tid];
  float m = block_max(v, red);
  float e = __expf(v - m);
  float sum = block_sum(e, red);
  float a = e / sum;
  int h = blk & 7, bj = blk >> 3, j = bj & 127, b = bj >> 7;
  a16[((b * kH + h) * 128 + j) * 128 + tid] = (_Float16)a;
}

// Aggregation per (b,h): o[j,c] = sum_i a[j,i] * xl[i,c]  (128x128 @ 128x16 WMMA)
__global__ void __launch_bounds__(256) k_attn(const _Float16* __restrict__ a16,
                                              const _Float16* __restrict__ xl16,
                                              const float* __restrict__ cb,
                                              _Float16* __restrict__ o16) {
  __shared__ _Float16 sAm[128 * 128];  // a[j][i]
  __shared__ _Float16 sBt[16 * 128];   // xl^T [c][i]
  int bh = blockIdx.x, b = bh >> 3, h = bh & 7;
  int tid = threadIdx.x;
#if USE_TDM
  if ((tid >> 5) == 0) tdm_copy_to_lds(sAm, a16 + bh * 16384, 128 * 128 * 2);
#else
  for (int idx = tid; idx < 128 * 128; idx += 256) sAm[idx] = a16[bh * 16384 + idx];
#endif
  for (int idx = tid; idx < 16 * 128; idx += 256) {
    int c = idx >> 7, i = idx & 127;
    sBt[idx] = xl16[(b * 128 + i) * 128 + h * 16 + c];
  }
  tdm_wait();
  __syncthreads();
  int w = tid >> 5;                     // wave owns row tile j in [16w, 16w+16)
  v8f acc = {0.f, 0.f, 0.f, 0.f, 0.f, 0.f, 0.f, 0.f};
#pragma unroll
  for (int ks = 0; ks < 4; ++ks) {
    v16h a = frag_a_from_lds(sAm + (w * 16) * 128, ks * 32);
    v16h bf = frag_b_from_lds_T(sBt, ks * 32);
    acc = __builtin_amdgcn_wmma_f32_16x16x32_f16(false, a, false, bf, (short)0, acc, false, false);
  }
  int lane = tid & 31, c = lane & 15, roff = (lane & 16) ? 8 : 0;
  float cbv = cb[h * 16 + c];
#pragma unroll
  for (int r = 0; r < 8; ++r) {
    int j = w * 16 + r + roff;
    o16[(b * 128 + j) * 128 + h * 16 + c] = (_Float16)(acc[r] + cbv);
  }
}

// Final: out[b] = (sum_n h[b,n,:]) @ out_w + out_b
__global__ void k_final(const float* __restrict__ h, const float* __restrict__ ow,
                        const float* __restrict__ ob, float* __restrict__ out) {
  __shared__ float red[8];
  int b = blockIdx.x, c = threadIdx.x;
  float acc = 0.f;
  for (int n = 0; n < 128; ++n) acc += h[(b * 128 + n) * 128 + c];
  float total = block_sum(acc * ow[c], red);
  if (c == 0) out[b] = total + ob[0];
}

// ---------------------------------------------------------------------------
extern "C" void kernel_launch(void* const* d_in, const int* in_sizes, int n_in,
                              void* d_out, int out_size, void* d_ws, size_t ws_size,
                              hipStream_t stream) {
  (void)in_sizes; (void)n_in; (void)out_size; (void)ws_size;
  const float* x      = (const float*)d_in[0];
  const int*   cat    = (const int*)  d_in[1];
  const float* emb    = (const float*)d_in[2];
  const float* mlp_w1 = (const float*)d_in[3];
  const float* mlp_b1 = (const float*)d_in[4];
  const float* ln1_g  = (const float*)d_in[5];
  const float* ln1_b  = (const float*)d_in[6];
  const float* mlp_w2 = (const float*)d_in[7];
  const float* mlp_b2 = (const float*)d_in[8];
  const float* ln2_g  = (const float*)d_in[9];
  const float* ln2_b  = (const float*)d_in[10];
  const float* Wl     = (const float*)d_in[11];
  const float* bl     = (const float*)d_in[12];
  const float* Wr     = (const float*)d_in[13];
  const float* We     = (const float*)d_in[14];
  const float* att    = (const float*)d_in[15];
  const float* cb     = (const float*)d_in[16];
  const float* pW     = (const float*)d_in[17];
  const float* pb     = (const float*)d_in[18];
  const float* lng    = (const float*)d_in[19];
  const float* lnb    = (const float*)d_in[20];
  const float* out_w  = (const float*)d_in[21];
  const float* out_b  = (const float*)d_in[22];
  float* out = (float*)d_out;

  const int rows = kBsz * kN;        // 2048
  char* base = (char*)d_ws; size_t off = 0;
  auto alloc = [&](size_t bytes) -> void* {
    void* p = base + off; off += (bytes + 255) & ~(size_t)255; return p;
  };
  _Float16* w2_16 = (_Float16*)alloc((size_t)kMat * 2);
  _Float16* Wl16  = (_Float16*)alloc((size_t)kL * kMat * 2);
  _Float16* Wr16  = (_Float16*)alloc((size_t)kL * kMat * 2);
  _Float16* We16  = (_Float16*)alloc((size_t)kL * kMat * 2);
  _Float16* pW16  = (_Float16*)alloc((size_t)kL * kMat * 2);
  _Float16* ef16  = (_Float16*)alloc((size_t)kN * kN * kHid * 2);
  _Float16* t16   = (_Float16*)alloc((size_t)rows * kHid * 2);
  float*    hbuf  = (float*)   alloc((size_t)rows * kHid * 4);
  _Float16* h16   = (_Float16*)alloc((size_t)rows * kHid * 2);
  float*    xlF   = (float*)   alloc((size_t)rows * kHid * 4);
  _Float16* xl16  = (_Float16*)alloc((size_t)rows * kHid * 2);
  float*    xrF   = (float*)   alloc((size_t)rows * kHid * 4);
  float*    e32   = (float*)   alloc((size_t)kN * kN * kHid * 4);
  float*    sbuf  = (float*)   alloc((size_t)rows * kH * kN * 4);
  _Float16* a16   = (_Float16*)alloc((size_t)rows * kH * kN * 2);
  _Float16* o16   = (_Float16*)alloc((size_t)rows * kHid * 2);

  // 1) weights -> f16; embedding gather
  k_cvt_f16<<<(kMat + 255) / 256, 256, 0, stream>>>(mlp_w2, w2_16, kMat);
  k_cvt_f16<<<(kL * kMat + 255) / 256, 256, 0, stream>>>(Wl, Wl16, kL * kMat);
  k_cvt_f16<<<(kL * kMat + 255) / 256, 256, 0, stream>>>(Wr, Wr16, kL * kMat);
  k_cvt_f16<<<(kL * kMat + 255) / 256, 256, 0, stream>>>(We, We16, kL * kMat);
  k_cvt_f16<<<(kL * kMat + 255) / 256, 256, 0, stream>>>(pW, pW16, kL * kMat);
  k_gather_embed<<<(kN * kN * kHid) / 256, 256, 0, stream>>>(emb, cat, ef16);

  // 2) input MLP
  k_input_mlp1<<<rows, 128, 0, stream>>>(x, mlp_w1, mlp_b1, ln1_g, ln1_b, t16);
  k_gemm_ln<2, false, false><<<rows / 32, 256, 0, stream>>>(
      t16, w2_16, mlp_b2, ln2_g, ln2_b, nullptr, hbuf, h16);

  // 3) GATv2 layers
  for (int l = 0; l < kL; ++l) {
    k_gemm<2, true, true, true><<<rows / 32, 256, 0, stream>>>(
        h16, Wl16 + l * kMat, bl + l * kHid, xlF, xl16);
    k_gemm<2, false, true, false><<<rows / 32, 256, 0, stream>>>(
        h16, Wr16 + l * kMat, nullptr, xrF, nullptr);
    k_gemm<4, false, true, false><<<(kN * kN) / 64, 256, 0, stream>>>(
        ef16, We16 + l * kMat, nullptr, e32, nullptr);
    k_score<<<rows, 256, 0, stream>>>(xlF, xrF, e32, att + l * kH * kC, sbuf);
    k_softmax<<<rows * kH, 128, 0, stream>>>(sbuf, a16);
    k_attn<<<kBsz * kH, 256, 0, stream>>>(a16, xl16, cb + l * kHid, o16);
    k_gemm_ln<2, true, true><<<rows / 32, 256, 0, stream>>>(
        o16, pW16 + l * kMat, pb + l * kHid, lng + l * kHid, lnb + l * kHid,
        hbuf, hbuf, h16);
  }

  // 4) readout
  k_final<<<kBsz, 128, 0, stream>>>(hbuf, out_w, out_b, out);
}
